// GAT_70849780515473
// MI455X (gfx1250) — compile-verified, matrix operand order checked by reading, not run on hardware
//
#include <hip/hip_runtime.h>
#include <hip/hip_bf16.h>
#include <stdint.h>

// ---------------------------------------------------------------------------
// Two-layer GAT forward for MI455X (gfx1250, wave32, WMMA + TDM).
// GEMMs run through v_wmma_f32_16x16x32_bf16 (bf16 in, f32 accumulate).
// GEMM1 tiles are staged into LDS by the Tensor Data Mover (tensor_load_to_lds,
// TENSORcnt). Softmax/aggregation stay fp32 and live in the 192MB L2.
// ---------------------------------------------------------------------------

typedef float        v8f   __attribute__((ext_vector_type(8)));
typedef __bf16       v16bf __attribute__((ext_vector_type(16)));
typedef unsigned int v8u   __attribute__((ext_vector_type(8)));
typedef unsigned int u32x4 __attribute__((ext_vector_type(4)));
typedef int          i32x4 __attribute__((ext_vector_type(4)));
typedef int          i32x8 __attribute__((ext_vector_type(8)));

#define NEG_SLOPE 0.2f

__device__ __forceinline__ unsigned short f2bf(float f) {
  unsigned u = __float_as_uint(f);
  unsigned rnd = 0x7fffu + ((u >> 16) & 1u);   // round-to-nearest-even
  return (unsigned short)((u + rnd) >> 16);
}

__device__ __forceinline__ void atomicMaxF(float* addr, float val) {
  unsigned* ua = (unsigned*)addr;
  unsigned old = *ua;
  while (__uint_as_float(old) < val) {
    unsigned assumed = old;
    old = atomicCAS(ua, assumed, __float_as_uint(val));
    if (old == assumed) break;
  }
}

// TDM: async 2D bf16 tile load Global->LDS via Tensor DMA Descriptor (D#).
// D# group0: [1:0]=count=1, [63:32]=lds_addr, [120:64]=global_addr, [127:126]=type=2
// D# group1: [17:16]=data_size(1=2B), [79:48]=tensor_dim0, [111:80]=tensor_dim1,
//            [127:112]=tile_dim0, [143:128]=tile_dim1, [207:160]=tensor_dim0_stride
// OOB reads (row >= tensor_dim1) return zero -> hardware zero-fill of edge tiles.
__device__ __forceinline__ void tdm_load_tile_2d(unsigned lds_off, const void* gsrc,
                                                 unsigned tensor_d0, unsigned tensor_d1,
                                                 unsigned tile_d0, unsigned tile_d1,
                                                 unsigned stride_d0) {
  unsigned long long ga = (unsigned long long)(uintptr_t)gsrc;
  u32x4 g0;
  g0.x = 1u;                                              // count=1 (valid)
  g0.y = lds_off;                                         // lds_addr (bytes)
  g0.z = (unsigned)(ga & 0xffffffffu);                    // global_addr[31:0]
  g0.w = (unsigned)((ga >> 32) & 0x01ffffffu) | (2u << 30); // addr[56:32] | type=2
  i32x8 g1;
  g1[0] = (int)(1u << 16);                                // data_size=2B; wg_mask=0
  g1[1] = (int)((tensor_d0 & 0xffffu) << 16);             // tensor_dim0[15:0]
  g1[2] = (int)(((tensor_d0 >> 16) & 0xffffu) |
                ((tensor_d1 & 0xffffu) << 16));           // dim0 hi | dim1 lo
  g1[3] = (int)(((tensor_d1 >> 16) & 0xffffu) |
                ((tile_d0 & 0xffffu) << 16));             // dim1 hi | tile_dim0
  g1[4] = (int)(tile_d1 & 0xffffu);                       // tile_dim1 (tile_dim2=0)
  g1[5] = (int)stride_d0;                                 // tensor_dim0_stride lo32
  g1[6] = 0;                                              // stride hi | dim1_stride lo
  g1[7] = 0;
  i32x4 z4 = {0, 0, 0, 0};
#if defined(__clang_major__) && (__clang_major__ >= 23)
  i32x8 z8 = {0, 0, 0, 0, 0, 0, 0, 0};
  __builtin_amdgcn_tensor_load_to_lds(g0, g1, z4, z4, z8, 0);
#else
  __builtin_amdgcn_tensor_load_to_lds(g0, g1, z4, z4, 0);
#endif
}

// ---------------- precision conversion kernels ----------------
__global__ void k_f32_to_bf16(const float* __restrict__ src,
                              unsigned short* __restrict__ dst, long long n) {
  long long i = (long long)blockIdx.x * blockDim.x + threadIdx.x;
  if (i < n) dst[i] = f2bf(src[i]);
}

// W1 [256,512] row-major -> W1T [512 out][256 k] bf16 (column-major of W1).
__global__ void k_w1_transpose(const float* __restrict__ W1,
                               unsigned short* __restrict__ W1T) {
  int i = blockIdx.x * blockDim.x + threadIdx.x;
  if (i >= 512 * 256) return;
  int o = i >> 8, k = i & 255;
  W1T[i] = f2bf(W1[(size_t)k * 512 + o]);
}

__global__ void k_zero_f32(float* __restrict__ p, long long n) {
  long long i = (long long)blockIdx.x * blockDim.x + threadIdx.x;
  if (i < n) p[i] = 0.f;
}

// ---------------- GEMM1: h1[N,512] = Xb[N,256] @ W1  (WMMA bf16 + TDM) ----
// Block = 128 threads = 4 waves. Block tile 64(M) x 64(N), K looped by 32.
// Wave 0 drives the TDM for both tiles; all 4 waves consume via ds_load.
__global__ __launch_bounds__(128) void k_gemm1_wmma(
    const unsigned short* __restrict__ Xb, const unsigned short* __restrict__ W1T,
    float* __restrict__ h1, int N) {
  __shared__ alignas(32) unsigned short As[64 * 32];   // 64 rows x 32 k, bf16
  __shared__ alignas(32) unsigned short Bs[64 * 32];   // 64 cols x 32 k, bf16
  const int tid = threadIdx.x;
  const int wave = tid >> 5, lane = tid & 31;
  const int m0 = blockIdx.x * 64;
  const int n0 = blockIdx.y * 64;
  v8f acc[4] = {};

  const unsigned asOff = (unsigned)(uintptr_t)(void*)As;   // LDS byte offset
  const unsigned bsOff = (unsigned)(uintptr_t)(void*)Bs;
  const unsigned remRows = (unsigned)(N - m0);             // OOB rows zero-filled

  for (int k0 = 0; k0 < 256; k0 += 32) {
    if (wave == 0) {
      // A tile: 64 rows x 32 k of Xb starting at (m0, k0); row stride 256 elems.
      tdm_load_tile_2d(asOff, Xb + (size_t)m0 * 256 + k0, 256u, remRows, 32u, 64u, 256u);
      // B tile: 64 output-cols x 32 k of W1T starting at (n0, k0).
      tdm_load_tile_2d(bsOff, W1T + (size_t)n0 * 256 + k0, 256u, 64u, 32u, 64u, 256u);
      __builtin_amdgcn_s_wait_tensorcnt(0);
    }
    __syncthreads();
    // A fragment: lanes 0-15 -> K[0:16), lanes 16-31 -> K[16:32) of row lane&15
    const int arow = 16 * wave + (lane & 15);
    const int kh = (lane >> 4) * 16;
    v8u araw = *(const v8u*)(As + arow * 32 + kh);
    v16bf af = __builtin_bit_cast(v16bf, araw);
#pragma unroll
    for (int j = 0; j < 4; ++j) {
      v8u braw = *(const v8u*)(Bs + (j * 16 + (lane & 15)) * 32 + kh);
      v16bf bf = __builtin_bit_cast(v16bf, braw);
      acc[j] = __builtin_amdgcn_wmma_f32_16x16x32_bf16(
          false, af, false, bf, (short)0, acc[j], false, false);
    }
    __syncthreads();
  }
  // C layout: VGPR r -> M = r + 8*(lane>=16); N = lane&15 (+16*j)
  const int rbase = m0 + 16 * wave + 8 * (lane >> 4);
  const int cbase = n0 + (lane & 15);
#pragma unroll
  for (int j = 0; j < 4; ++j)
#pragma unroll
    for (int r = 0; r < 8; ++r) {
      int row = rbase + r;
      if (row < N) h1[(size_t)row * 512 + cbase + j * 16] = acc[j][r];
    }
}

// ---------------- per-node attention logits, layer 1 ----------------------
__global__ void k_node_logits1(const float* __restrict__ h1,
                               const float* __restrict__ a_src,
                               const float* __restrict__ a_dst,
                               float* __restrict__ alS, float* __restrict__ alD,
                               float* __restrict__ m1, float* __restrict__ dd1,
                               int N) {
  int idx = blockIdx.x * blockDim.x + threadIdx.x;
  if (idx >= N * 8) return;
  int n = idx >> 3, h = idx & 7;
  const float* hp = h1 + (size_t)n * 512 + h * 64;
  const float* as = a_src + h * 64;
  const float* ad = a_dst + h * 64;
  float s = 0.f, d = 0.f;
#pragma unroll 8
  for (int c = 0; c < 64; ++c) { float v = hp[c]; s += v * as[c]; d += v * ad[c]; }
  alS[idx] = s; alD[idx] = d;
  m1[idx] = -1e30f; dd1[idx] = 0.f;
}

// ---------------- edge passes, layer 1 (segment softmax) ------------------
__global__ void k_edge_logits1(const long long* __restrict__ ei,
                               const float* __restrict__ alS, const float* __restrict__ alD,
                               float* __restrict__ logits, float* __restrict__ m1,
                               long long E, long long Ep) {
  long long idx = (long long)blockIdx.x * blockDim.x + threadIdx.x;
  if (idx >= Ep * 8) return;
  long long e = idx >> 3; int h = idx & 7;
  long long s = (e < E) ? ei[e] : (e - E);
  long long d = (e < E) ? ei[E + e] : (e - E);
  float l = alS[s * 8 + h] + alD[d * 8 + h];
  l = (l > 0.f) ? l : NEG_SLOPE * l;
  logits[idx] = l;
  atomicMaxF(&m1[d * 8 + h], l);
}

__global__ void k_edge_exp1(const long long* __restrict__ ei,
                            float* __restrict__ logits, const float* __restrict__ m1,
                            float* __restrict__ dd1, long long E, long long Ep) {
  long long idx = (long long)blockIdx.x * blockDim.x + threadIdx.x;
  if (idx >= Ep * 8) return;
  long long e = idx >> 3; int h = idx & 7;
  long long d = (e < E) ? ei[E + e] : (e - E);
  float ex = __expf(logits[idx] - m1[d * 8 + h]);
  logits[idx] = ex;
  atomicAdd(&dd1[d * 8 + h], ex);
}

// Aggregate: agg[dst, c] += alpha * h1[src, c]; everything L2-resident.
__global__ void k_edge_agg1(const long long* __restrict__ ei,
                            const float* __restrict__ logits, const float* __restrict__ dd1,
                            const float* __restrict__ h1, float* __restrict__ agg,
                            long long E, long long Ep) {
  long long idx = (long long)blockIdx.x * blockDim.x + threadIdx.x;
  if (idx >= Ep * 512) return;
  long long e = idx >> 9; int c = (int)(idx & 511); int h = c >> 6;
  long long s = (e < E) ? ei[e] : (e - E);
  long long d = (e < E) ? ei[E + e] : (e - E);
  float alpha = logits[e * 8 + h] / (dd1[d * 8 + h] + 1e-16f);
  atomicAdd(&agg[d * 512 + c], alpha * h1[s * 512 + c]);
}

// out1 = elu(agg + b1); store directly as bf16 (only consumer is GEMM2).
__global__ void k_act_conv(const float* __restrict__ agg, const float* __restrict__ b1,
                           unsigned short* __restrict__ Ab, long long n) {
  long long i = (long long)blockIdx.x * blockDim.x + threadIdx.x;
  if (i >= n) return;
  int c = (int)(i & 511);
  float v = agg[i] + b1[c];
  v = (v > 0.f) ? v : (__expf(v) - 1.f);
  Ab[i] = f2bf(v);
}

// WcT [16 rows x 512 k] bf16: rows 0-3 = W2 cols; row4 = W2@a_src2; row5 = W2@a_dst2.
__global__ void k_build_wc(const float* __restrict__ W2, const float* __restrict__ aS2,
                           const float* __restrict__ aD2, unsigned short* __restrict__ WcT) {
  int t = blockIdx.x * blockDim.x + threadIdx.x;
  if (t >= 16 * 512) return;
  int row = t >> 9, k = t & 511;
  float v = 0.f;
  if (row < 4) v = W2[(size_t)k * 4 + row];
  else if (row == 4) { for (int o = 0; o < 4; ++o) v += W2[(size_t)k * 4 + o] * aS2[o]; }
  else if (row == 5) { for (int o = 0; o < 4; ++o) v += W2[(size_t)k * 4 + o] * aD2[o]; }
  WcT[row * 512 + k] = f2bf(v);
}

// ---------------- GEMM2 (fused logits): [N,512] @ WcT^T -> h2 + al2 -------
__global__ __launch_bounds__(256) void k_gemm2_wmma(
    const unsigned short* __restrict__ Ab, const unsigned short* __restrict__ WcT,
    float* __restrict__ h2, float* __restrict__ al2S, float* __restrict__ al2D, int N) {
  const int wave = threadIdx.x >> 5, lane = threadIdx.x & 31;
  const int m0 = blockIdx.x * 128 + wave * 16;
  const int arow = m0 + (lane & 15);
  const bool va = arow < N;
  const int kh = (lane >> 4) * 16;
  v8f acc = {};
  const unsigned short* ap = Ab + (size_t)(va ? arow : 0) * 512;
  const unsigned short* bp = WcT + (size_t)(lane & 15) * 512;
  for (int k0 = 0; k0 < 512; k0 += 32) {
    v8u araw = {};
    if (va) araw = *(const v8u*)(ap + k0 + kh);
    v8u braw = *(const v8u*)(bp + k0 + kh);
    if (k0 + 32 < 512) __builtin_prefetch(ap + k0 + 32 + kh, 0, 0);
    acc = __builtin_amdgcn_wmma_f32_16x16x32_bf16(
        false, __builtin_bit_cast(v16bf, araw),
        false, __builtin_bit_cast(v16bf, braw), (short)0, acc, false, false);
  }
  const int col = lane & 15;
#pragma unroll
  for (int r = 0; r < 8; ++r) {
    int row = m0 + 8 * (lane >> 4) + r;
    if (row < N) {
      float v = acc[r];
      if (col < 4)       h2[(size_t)row * 4 + col] = v;
      else if (col == 4) al2S[row] = v;
      else if (col == 5) al2D[row] = v;
    }
  }
}

// ---------------- layer-2 softmax / aggregation ---------------------------
__global__ void k_init2(float* __restrict__ m2, float* __restrict__ dd2,
                        float* __restrict__ agg2, int N) {
  int n = blockIdx.x * blockDim.x + threadIdx.x;
  if (n >= N) return;
  m2[n] = -1e30f; dd2[n] = 0.f;
  agg2[n * 4 + 0] = 0.f; agg2[n * 4 + 1] = 0.f;
  agg2[n * 4 + 2] = 0.f; agg2[n * 4 + 3] = 0.f;
}

__global__ void k_edge_logits2(const long long* __restrict__ ei,
                               const float* __restrict__ alS, const float* __restrict__ alD,
                               float* __restrict__ logit2, float* __restrict__ m2,
                               long long E, long long Ep) {
  long long e = (long long)blockIdx.x * blockDim.x + threadIdx.x;
  if (e >= Ep) return;
  long long s = (e < E) ? ei[e] : (e - E);
  long long d = (e < E) ? ei[E + e] : (e - E);
  float l = alS[s] + alD[d];
  l = (l > 0.f) ? l : NEG_SLOPE * l;
  logit2[e] = l;
  atomicMaxF(&m2[d], l);
}

__global__ void k_edge_exp2(const long long* __restrict__ ei,
                            float* __restrict__ logit2, const float* __restrict__ m2,
                            float* __restrict__ dd2, long long E, long long Ep) {
  long long e = (long long)blockIdx.x * blockDim.x + threadIdx.x;
  if (e >= Ep) return;
  long long d = (e < E) ? ei[E + e] : (e - E);
  float ex = __expf(logit2[e] - m2[d]);
  logit2[e] = ex;
  atomicAdd(&dd2[d], ex);
}

__global__ void k_edge_agg2(const long long* __restrict__ ei,
                            const float* __restrict__ logit2, const float* __restrict__ dd2,
                            const float* __restrict__ h2, float* __restrict__ agg2,
                            long long E, long long Ep) {
  long long idx = (long long)blockIdx.x * blockDim.x + threadIdx.x;
  if (idx >= Ep * 4) return;
  long long e = idx >> 2; int o = (int)(idx & 3);
  long long s = (e < E) ? ei[e] : (e - E);
  long long d = (e < E) ? ei[E + e] : (e - E);
  float alpha = logit2[e] / (dd2[d] + 1e-16f);
  atomicAdd(&agg2[d * 4 + o], alpha * h2[s * 4 + o]);
}

__global__ void k_final(const float* __restrict__ agg2, const float* __restrict__ b2,
                        float* __restrict__ out, int N) {
  int idx = blockIdx.x * blockDim.x + threadIdx.x;
  if (idx >= N * 4) return;
  out[idx] = agg2[idx] + b2[idx & 3];
}

// ---------------------------------------------------------------------------
extern "C" void kernel_launch(void* const* d_in, const int* in_sizes, int n_in,
                              void* d_out, int out_size, void* d_ws, size_t ws_size,
                              hipStream_t stream) {
  const float*      x   = (const float*)d_in[0];
  const long long*  ei  = (const long long*)d_in[1];   // int64 edge_index [2,E]
  const float*      W1  = (const float*)d_in[2];
  const float*      aS1 = (const float*)d_in[3];
  const float*      aD1 = (const float*)d_in[4];
  const float*      b1  = (const float*)d_in[5];
  const float*      W2  = (const float*)d_in[6];
  const float*      aS2 = (const float*)d_in[7];
  const float*      aD2 = (const float*)d_in[8];
  const float*      b2  = (const float*)d_in[9];
  float*            out = (float*)d_out;

  const long long N  = in_sizes[0] / 256;
  const long long E  = in_sizes[1] / 2;
  const long long Ep = E + N;

  // workspace carve-out
  char* base = (char*)d_ws;
  size_t off = 0;
  auto alloc = [&](size_t bytes) -> char* {
    off = (off + 255) & ~(size_t)255;
    char* p = base + off; off += bytes; return p;
  };
  unsigned short* Xb     = (unsigned short*)alloc((size_t)N * 256 * 2);
  unsigned short* W1T    = (unsigned short*)alloc((size_t)512 * 256 * 2);
  float*          h1     = (float*)alloc((size_t)N * 512 * 4);
  float*          alS1   = (float*)alloc((size_t)N * 8 * 4);
  float*          alD1   = (float*)alloc((size_t)N * 8 * 4);
  float*          m1     = (float*)alloc((size_t)N * 8 * 4);
  float*          dd1    = (float*)alloc((size_t)N * 8 * 4);
  float*          lg1    = (float*)alloc((size_t)Ep * 8 * 4);
  float*          agg1   = (float*)alloc((size_t)N * 512 * 4);
  unsigned short* Ab     = (unsigned short*)alloc((size_t)N * 512 * 2);
  unsigned short* WcT    = (unsigned short*)alloc((size_t)16 * 512 * 2);
  float*          h2s    = (float*)alloc((size_t)N * 4 * 4);
  float*          al2S   = (float*)alloc((size_t)N * 4);
  float*          al2D   = (float*)alloc((size_t)N * 4);
  float*          m2     = (float*)alloc((size_t)N * 4);
  float*          dd2    = (float*)alloc((size_t)N * 4);
  float*          lg2    = (float*)alloc((size_t)Ep * 4);
  float*          agg2   = (float*)alloc((size_t)N * 4 * 4);
  (void)ws_size; (void)n_in; (void)out_size;

  auto cdiv = [](long long a, long long b) { return (int)((a + b - 1) / b); };

  k_f32_to_bf16 <<<cdiv(N * 256, 256), 256, 0, stream>>>(x, Xb, N * 256);
  k_w1_transpose<<<cdiv(512 * 256, 256), 256, 0, stream>>>(W1, W1T);

  dim3 g1(cdiv(N, 64), 8);
  k_gemm1_wmma  <<<g1, 128, 0, stream>>>(Xb, W1T, h1, (int)N);

  k_node_logits1<<<cdiv(N * 8, 256), 256, 0, stream>>>(h1, aS1, aD1, alS1, alD1, m1, dd1, (int)N);
  k_zero_f32    <<<cdiv(N * 512, 256), 256, 0, stream>>>(agg1, N * 512);

  k_edge_logits1<<<cdiv(Ep * 8, 256), 256, 0, stream>>>(ei, alS1, alD1, lg1, m1, E, Ep);
  k_edge_exp1   <<<cdiv(Ep * 8, 256), 256, 0, stream>>>(ei, lg1, m1, dd1, E, Ep);
  k_edge_agg1   <<<cdiv(Ep * 512, 256), 256, 0, stream>>>(ei, lg1, dd1, h1, agg1, E, Ep);

  k_act_conv    <<<cdiv(N * 512, 256), 256, 0, stream>>>(agg1, b1, Ab, N * 512);
  k_build_wc    <<<cdiv(16 * 512, 256), 256, 0, stream>>>(W2, aS2, aD2, WcT);
  k_gemm2_wmma  <<<cdiv(N, 128), 256, 0, stream>>>(Ab, WcT, h2s, al2S, al2D, (int)N);

  k_init2       <<<cdiv(N, 256), 256, 0, stream>>>(m2, dd2, agg2, (int)N);
  k_edge_logits2<<<cdiv(Ep, 256), 256, 0, stream>>>(ei, al2S, al2D, lg2, m2, E, Ep);
  k_edge_exp2   <<<cdiv(Ep, 256), 256, 0, stream>>>(ei, lg2, m2, dd2, E, Ep);
  k_edge_agg2   <<<cdiv(Ep * 4, 256), 256, 0, stream>>>(ei, lg2, dd2, h2s, agg2, E, Ep);
  k_final       <<<cdiv(N * 4, 256), 256, 0, stream>>>(agg2, b2, out, (int)N);
}
